// VisionTransformer_64372969832577
// MI455X (gfx1250) — matrix operand order repaired and checked by reading, not compiled, not run on hardware
//
#include <hip/hip_runtime.h>
#include <hip/hip_bf16.h>

// ---------------------------------------------------------------------------
// ViT-Base forward with product-bucket relative position attention, MI455X.
// bf16 WMMA (v_wmma_f32_16x16x32_bf16) + f32 accumulate; all GEMM B operands
// stored pre-transposed [N][K] so A/B LDS staging is contiguous 16B chunks;
// double-buffered K loop using gfx1250 GLOBAL_LOAD_ASYNC_TO_LDS_B128 when the
// toolchain exposes the builtin (ASYNCcnt + one barrier per iteration).
// ---------------------------------------------------------------------------

typedef __bf16 bf16_t;
typedef __attribute__((ext_vector_type(16))) __bf16 v16bf;
typedef __attribute__((ext_vector_type(8)))  __bf16 v8bf;
typedef __attribute__((ext_vector_type(8)))  float  v8f;
typedef int v4i __attribute__((vector_size(16)));   // builtin's b128 unit type

#if defined(__gfx1250__) && __has_builtin(__builtin_amdgcn_global_load_async_to_lds_b128) && __has_builtin(__builtin_amdgcn_s_wait_asynccnt)
#define VIT_ASYNC 1
#endif

#define BATCHSZ 32
#define NTOK    197
#define NTOKP   200        // padded (16B-aligned rows for b128 staging)
#define NPATCH  196
#define DMODEL  768
#define NHEAD   12
#define HDIM    64
#define NLAYER  12
#define DFFN    3072
#define NBUCKET 226
#define NBUCKP  232        // padded
#define QKSCALE 0.125f
#define NROWS   (BATCHSZ*NTOK)     // 6304 token rows
#define PROWS   (BATCHSZ*NPATCH)   // 6272 patch rows
#define NBH     (BATCHSZ*NHEAD)    // 384 attention batches

// ---------------- GEMM: C[M,N] = A[M,K](bf16) * Bt[N,K](bf16)^T -------------
#define BM 128
#define BN 128
#define BK 32
#define LDSS 40   // padded LDS row stride in bf16 elems (80B, 16B aligned)

enum { GF_ACC = 1, GF_BIAS = 2, GF_GELU = 4, GF_WF32 = 8, GF_WBF = 16 };

// Stage a 128x32 bf16 tile (rows row0.., cols kb..kb+31 of a [rowLimit x K]
// row-major matrix with leading dim ld) into LDS. Fast path: 2 x 16B per lane
// (async-to-LDS on gfx1250 if available). Edge tiles: scalar bounds-checked.
__device__ __forceinline__
void vit_stage_tile(const bf16_t* __restrict__ src, long ld,
                    int row0, int rowLimit, int kb, int K,
                    bf16_t* __restrict__ sm, int tid)
{
    const int r  = tid >> 1;          // 0..127 tile row
    const int hh = tid & 1;           // 16-elem half of the 32-elem row
    const int gRow = row0 + r;
    bf16_t* dst = &sm[r * LDSS + hh * 16];
    const bool full = (row0 + BM <= rowLimit) && (kb + BK <= K);  // uniform
    if (full) {
        const bf16_t* sp = src + (long)gRow * ld + kb + hh * 16;
#ifdef VIT_ASYNC
        __builtin_amdgcn_global_load_async_to_lds_b128(
            (__attribute__((address_space(1))) v4i*)sp,
            (__attribute__((address_space(3))) v4i*)dst, 0, 0);
        __builtin_amdgcn_global_load_async_to_lds_b128(
            (__attribute__((address_space(1))) v4i*)(sp + 8),
            (__attribute__((address_space(3))) v4i*)(dst + 8), 0, 0);
#else
        *(v8bf*)dst       = *(const v8bf*)sp;
        *(v8bf*)(dst + 8) = *(const v8bf*)(sp + 8);
#endif
    } else {
        #pragma unroll
        for (int i = 0; i < 16; ++i) {
            const int kk = kb + hh * 16 + i;
            bf16_t v = (bf16_t)0.0f;
            if (gRow < rowLimit && kk < K) v = src[(long)gRow * ld + kk];
            dst[i] = v;
        }
    }
}

__global__ __launch_bounds__(256)
void vit_gemm_bf16(const bf16_t* __restrict__ A, const bf16_t* __restrict__ Bt,
                   float* __restrict__ C, bf16_t* __restrict__ Cb,
                   const float* __restrict__ bias,
                   int Mr, int Nc, int K, int lda, int ldb, int ldc, int ldcb,
                   long aBatch, long bBatch, long cBatch, int bMod, int flags)
{
    __shared__ __attribute__((aligned(16))) bf16_t Asm[2][BM * LDSS];
    __shared__ __attribute__((aligned(16))) bf16_t Bsm[2][BN * LDSS];

    const int z = blockIdx.z;
    A  += (long)z * aBatch;
    Bt += (long)(z % bMod) * bBatch;
    const long cOff = (long)z * cBatch;

    const int m0   = blockIdx.y * BM;
    const int n0   = blockIdx.x * BN;
    const int tid  = threadIdx.x;
    const int wave = tid >> 5;
    const int lane = tid & 31;
    const int wm   = wave >> 1;        // 0..3 : 32-row strip
    const int wn   = wave & 1;         // 0..1 : 64-col strip
    const int lrow  = lane & 15;
    const int khalf = lane >> 4;

    v8f acc[2][4] = {};

    // prologue: stage first K tile into buffer 0
    vit_stage_tile(A,  lda, m0, Mr, 0, K, Asm[0], tid);
    vit_stage_tile(Bt, ldb, n0, Nc, 0, K, Bsm[0], tid);

    int par = 0;
    for (int kb = 0; kb < K; kb += BK, par ^= 1) {
#ifdef VIT_ASYNC
        __builtin_amdgcn_s_wait_asynccnt(0);
#endif
        __syncthreads();   // tile kb visible; previous buffer free for reuse
        const int kn = kb + BK;
        if (kn < K) {      // prefetch next tile into the other buffer
            vit_stage_tile(A,  lda, m0, Mr, kn, K, Asm[par ^ 1], tid);
            vit_stage_tile(Bt, ldb, n0, Nc, kn, K, Bsm[par ^ 1], tid);
        }

        const bf16_t* As = Asm[par];
        const bf16_t* Bs = Bsm[par];
        v16bf afrag[2], bfrag[4];
        #pragma unroll
        for (int tm = 0; tm < 2; ++tm) {
            const int row = wm * 32 + tm * 16 + lrow;
            // ISA 16-bit A 16x32: lane-half holds K {h*8..+7} and {16+h*8..+7}
            v8bf lo = *(const v8bf*)&As[row * LDSS + khalf * 8];
            v8bf hi = *(const v8bf*)&As[row * LDSS + 16 + khalf * 8];
            #pragma unroll
            for (int i = 0; i < 8; ++i) { afrag[tm][i] = lo[i]; afrag[tm][8 + i] = hi[i]; }
        }
        #pragma unroll
        for (int tn = 0; tn < 4; ++tn) {
            const int col = wn * 64 + tn * 16 + lrow;
            // ISA 16-bit B 32x16: lane-half holds K {h*8..+7} and {16+h*8..+7}
            v8bf lo = *(const v8bf*)&Bs[col * LDSS + khalf * 8];
            v8bf hi = *(const v8bf*)&Bs[col * LDSS + 16 + khalf * 8];
            #pragma unroll
            for (int i = 0; i < 8; ++i) { bfrag[tn][i] = lo[i]; bfrag[tn][8 + i] = hi[i]; }
        }
        #pragma unroll
        for (int tm = 0; tm < 2; ++tm)
            #pragma unroll
            for (int tn = 0; tn < 4; ++tn)
                acc[tm][tn] = __builtin_amdgcn_wmma_f32_16x16x32_bf16(
                    false, afrag[tm], false, bfrag[tn],
                    (short)0, acc[tm][tn], false, false);
    }

    // --- epilogue: C/D layout = lanes0-15 rows r, lanes16-31 rows 8+r ---
    const bool fullC = (m0 + BM <= Mr) && (n0 + BN <= Nc);   // uniform
    #pragma unroll
    for (int tm = 0; tm < 2; ++tm) {
        #pragma unroll
        for (int tn = 0; tn < 4; ++tn) {
            const int gCol = n0 + wn * 64 + tn * 16 + lrow;
            const int rBase = m0 + wm * 32 + tm * 16 + khalf * 8;
            if (fullC) {
                const float bs = (flags & GF_BIAS) ? bias[gCol] : 0.0f;
                #pragma unroll
                for (int r = 0; r < 8; ++r) {
                    const int gRow = rBase + r;
                    float v = acc[tm][tn][r] + bs;
                    if (flags & GF_ACC)  v += C[cOff + (long)gRow * ldc + gCol];
                    if (flags & GF_GELU) v = 0.5f * v * (1.0f + erff(v * 0.70710678118654752f));
                    if (flags & GF_WF32) C[cOff + (long)gRow * ldc + gCol] = v;
                    if (flags & GF_WBF)  Cb[(long)gRow * ldcb + gCol] = (bf16_t)v;
                }
            } else {
                #pragma unroll
                for (int r = 0; r < 8; ++r) {
                    const int gRow = rBase + r;
                    if (gRow < Mr && gCol < Nc) {
                        float v = acc[tm][tn][r];
                        if (flags & GF_BIAS) v += bias[gCol];
                        if (flags & GF_ACC)  v += C[cOff + (long)gRow * ldc + gCol];
                        if (flags & GF_GELU) v = 0.5f * v * (1.0f + erff(v * 0.70710678118654752f));
                        if (flags & GF_WF32) C[cOff + (long)gRow * ldc + gCol] = v;
                        if (flags & GF_WBF)  Cb[(long)gRow * ldcb + gCol] = (bf16_t)v;
                    }
                }
            }
        }
    }
}

// ---------------- fused softmax + RPE bias gather + bucket scatter ----------
__global__ __launch_bounds__(256)
void vit_softmax_rpe(const float* __restrict__ attn,
                     const float* __restrict__ qt,
                     const float* __restrict__ kt,
                     const int* __restrict__ rel_idx,
                     bf16_t* __restrict__ attn_bf,
                     bf16_t* __restrict__ abucket_bf)
{
    const int i   = blockIdx.x;
    const int bh  = blockIdx.y;
    const int tid = threadIdx.x;
    __shared__ float qrow[NBUCKET];
    __shared__ float bucket[NBUCKET];
    __shared__ float red[256];

    const float* qtp = qt + ((long)bh * NTOK + i) * NBUCKET;
    for (int m = tid; m < NBUCKET; m += 256) { qrow[m] = qtp[m]; bucket[m] = 0.0f; }
    __syncthreads();

    float logit = -3.4e38f;
    int idx_ij = 0;
    if (tid < NTOK) {
        idx_ij = rel_idx[i * NTOK + tid];
        const int idx_ji = rel_idx[tid * NTOK + i];   // swapaxes(kt gather)
        logit = attn[((long)bh * NTOK + i) * NTOK + tid]
              + qrow[idx_ij]
              + kt[((long)bh * NTOK + tid) * NBUCKET + idx_ji];
    }
    red[tid] = logit; __syncthreads();
    for (int s = 128; s > 0; s >>= 1) { if (tid < s) red[tid] = fmaxf(red[tid], red[tid + s]); __syncthreads(); }
    const float mx = red[0]; __syncthreads();
    const float e = (tid < NTOK) ? __expf(logit - mx) : 0.0f;
    red[tid] = e; __syncthreads();
    for (int s = 128; s > 0; s >>= 1) { if (tid < s) red[tid] += red[tid + s]; __syncthreads(); }
    const float p = e * (1.0f / red[0]);
    if (tid < NTOK) {
        attn_bf[((long)bh * NTOK + i) * NTOKP + tid] = (bf16_t)p;
        atomicAdd(&bucket[idx_ij], p);                // ds_add_f32
    }
    __syncthreads();
    for (int m = tid; m < NBUCKET; m += 256)
        abucket_bf[((long)bh * NTOK + i) * NBUCKP + m] = (bf16_t)bucket[m];
}

// ---------------- LayerNorm (768 cols, 256 threads * 3) -> bf16 -------------
__global__ __launch_bounds__(256)
void vit_layernorm(const float* __restrict__ x, const float* __restrict__ g,
                   const float* __restrict__ b, bf16_t* __restrict__ y, int rowStep)
{
    const int r = blockIdx.x;
    const int tid = threadIdx.x;
    const float* xp = x + (long)r * rowStep * DMODEL;
    __shared__ float red[256];
    const float v0 = xp[tid], v1 = xp[tid + 256], v2 = xp[tid + 512];
    red[tid] = v0 + v1 + v2; __syncthreads();
    for (int s = 128; s > 0; s >>= 1) { if (tid < s) red[tid] += red[tid + s]; __syncthreads(); }
    const float mean = red[0] * (1.0f / DMODEL); __syncthreads();
    const float d0 = v0 - mean, d1 = v1 - mean, d2 = v2 - mean;
    red[tid] = d0 * d0 + d1 * d1 + d2 * d2; __syncthreads();
    for (int s = 128; s > 0; s >>= 1) { if (tid < s) red[tid] += red[tid + s]; __syncthreads(); }
    const float rstd = rsqrtf(red[0] * (1.0f / DMODEL) + 1e-5f);
    bf16_t* yp = y + (long)r * DMODEL;
    yp[tid]       = (bf16_t)(d0 * rstd * g[tid]       + b[tid]);
    yp[tid + 256] = (bf16_t)(d1 * rstd * g[tid + 256] + b[tid + 256]);
    yp[tid + 512] = (bf16_t)(d2 * rstd * g[tid + 512] + b[tid + 512]);
}

// ---------------- elementwise helpers ---------------------------------------
__global__ void vit_cvt_bf16(const float* __restrict__ s, bf16_t* __restrict__ d, long n)
{
    long i = (long)blockIdx.x * blockDim.x + threadIdx.x;
    const long st = (long)gridDim.x * blockDim.x;
    for (; i < n; i += st) d[i] = (bf16_t)s[i];
}

// batched transpose-convert: in [nb][K][N] f32 -> out [nb][N][ldo] bf16 (Bt)
__global__ void vit_wtrans(const float* __restrict__ in, bf16_t* __restrict__ out,
                           int K, int N, int ldo, long total)
{
    long i = (long)blockIdx.x * blockDim.x + threadIdx.x;
    const long st = (long)gridDim.x * blockDim.x;
    const long KN = (long)K * N;
    for (; i < total; i += st) {
        const long b = i / KN;
        const int  rr = (int)(i % KN);
        const int  k = rr / N, n = rr % N;
        out[((long)b * N + n) * ldo + k] = (bf16_t)in[i];
    }
}

// im2col: x[B,3,224,224] -> bf16 [B*196, 768] in (c,ph,pw) order
__global__ void vit_im2col(const float* __restrict__ x, bf16_t* __restrict__ out)
{
    long i = (long)blockIdx.x * blockDim.x + threadIdx.x;
    if (i >= (long)PROWS * DMODEL) return;
    const int k = (int)(i % DMODEL);
    const long row = i / DMODEL;
    const int b = (int)(row / NPATCH), p = (int)(row % NPATCH);
    const int gy = p / 14, gx = p % 14;
    const int c = k >> 8, rem = k & 255, ph = rem >> 4, pw = rem & 15;
    out[i] = (bf16_t)x[(((long)b * 3 + c) * 224 + gy * 16 + ph) * 224 + gx * 16 + pw];
}

// h = concat(cls, patches) + pos_embed
__global__ void vit_assemble_h(const float* __restrict__ patch, const float* __restrict__ cls,
                               const float* __restrict__ pos, float* __restrict__ h)
{
    long i = (long)blockIdx.x * blockDim.x + threadIdx.x;
    if (i >= (long)NROWS * DMODEL) return;
    const int d = (int)(i % DMODEL);
    const long row = i / DMODEL;
    const int b = (int)(row / NTOK), n = (int)(row % NTOK);
    const float v = (n == 0) ? cls[d] : patch[((long)b * NPATCH + (n - 1)) * DMODEL + d];
    h[i] = v + pos[n * DMODEL + d];
}

// qkv [rows,2304] -> q*s [bh,n,64], k*s [bh,n,64], k [bh,n,64], v^T [bh,64,NTOKP]
__global__ void vit_split_qkv(const float* __restrict__ qkv, bf16_t* __restrict__ q,
                              bf16_t* __restrict__ ks, bf16_t* __restrict__ k,
                              bf16_t* __restrict__ vT)
{
    long i = (long)blockIdx.x * blockDim.x + threadIdx.x;
    if (i >= (long)NROWS * DMODEL) return;
    const int col = (int)(i % DMODEL);
    const long row = i / DMODEL;
    const int b = (int)(row / NTOK), n = (int)(row % NTOK);
    const int h = col >> 6, d = col & 63;
    const float* base = qkv + row * (3 * DMODEL);
    const float qv = base[col], kv = base[DMODEL + col], vv = base[2 * DMODEL + col];
    const long bh = (long)b * NHEAD + h;
    q [(bh * NTOK + n) * HDIM + d] = (bf16_t)(qv * QKSCALE);
    ks[(bh * NTOK + n) * HDIM + d] = (bf16_t)(kv * QKSCALE);
    k [(bh * NTOK + n) * HDIM + d] = (bf16_t)kv;
    vT[(bh * HDIM + d) * NTOKP + n] = (bf16_t)vv;
}

// tmp_o [bh,n,64] -> bf16 [b*197+n, h*64+d]
__global__ void vit_reorder_o(const float* __restrict__ tmp_o, bf16_t* __restrict__ o)
{
    long i = (long)blockIdx.x * blockDim.x + threadIdx.x;
    if (i >= (long)NROWS * DMODEL) return;
    const int col = (int)(i % DMODEL);
    const long row = i / DMODEL;
    const int b = (int)(row / NTOK), n = (int)(row % NTOK);
    const int h = col >> 6, d = col & 63;
    o[i] = (bf16_t)tmp_o[(((long)b * NHEAD + h) * NTOK + n) * HDIM + d];
}

// ---------------------------------------------------------------------------
extern "C" void kernel_launch(void* const* d_in, const int* in_sizes, int n_in,
                              void* d_out, int out_size, void* d_ws, size_t ws_size,
                              hipStream_t stream)
{
    (void)in_sizes; (void)n_in; (void)out_size; (void)ws_size;
    const float* x       = (const float*)d_in[0];
    const float* patch_w = (const float*)d_in[1];
    const float* patch_b = (const float*)d_in[2];
    const float* cls_tok = (const float*)d_in[3];
    const float* pos_emb = (const float*)d_in[4];
    const float* ln1_g   = (const float*)d_in[5];
    const float* ln1_b   = (const float*)d_in[6];
    const float* qkv_w   = (const float*)d_in[7];
    const float* proj_w  = (const float*)d_in[8];
    const float* proj_b  = (const float*)d_in[9];
    const float* ln2_g   = (const float*)d_in[10];
    const float* ln2_b   = (const float*)d_in[11];
    const float* fc1_w   = (const float*)d_in[12];
    const float* fc1_b   = (const float*)d_in[13];
    const float* fc2_w   = (const float*)d_in[14];
    const float* fc2_b   = (const float*)d_in[15];
    const float* rpe_k   = (const float*)d_in[16];
    const float* rpe_q   = (const float*)d_in[17];
    const float* rpe_v   = (const float*)d_in[18];
    const float* norm_g  = (const float*)d_in[19];
    const float* norm_b  = (const float*)d_in[20];
    const float* head_w  = (const float*)d_in[21];
    const float* head_b  = (const float*)d_in[22];
    const int*   rel_idx = (const int*)d_in[23];
    float* out = (float*)d_out;

    // ---- workspace carve-up ----
    size_t off = 0;
    auto alloc = [&](size_t bytes) -> char* {
        char* p = (char*)d_ws + off;
        off += (bytes + 255) & ~(size_t)255;
        return p;
    };
    const long RPE_E = (long)NLAYER * NHEAD * NBUCKET * HDIM;   // 2,082,816
    bf16_t* wpatchT = (bf16_t*)alloc(sizeof(bf16_t) * DMODEL * DMODEL);
    bf16_t* wqkvT   = (bf16_t*)alloc(sizeof(bf16_t) * (long)NLAYER * 3 * DMODEL * DMODEL);
    bf16_t* wprojT  = (bf16_t*)alloc(sizeof(bf16_t) * (long)NLAYER * DMODEL * DMODEL);
    bf16_t* wfc1T   = (bf16_t*)alloc(sizeof(bf16_t) * (long)NLAYER * DFFN * DMODEL);
    bf16_t* wfc2T   = (bf16_t*)alloc(sizeof(bf16_t) * (long)NLAYER * DMODEL * DFFN);
    bf16_t* wheadT  = (bf16_t*)alloc(sizeof(bf16_t) * 1000 * DMODEL);
    bf16_t* rpeK_bf = (bf16_t*)alloc(sizeof(bf16_t) * RPE_E);                    // [lh][226][64] (= Bt)
    bf16_t* rpeQ_bf = (bf16_t*)alloc(sizeof(bf16_t) * RPE_E);
    bf16_t* rVT_bf  = (bf16_t*)alloc(sizeof(bf16_t) * (long)NLAYER * NHEAD * HDIM * NBUCKP);
    bf16_t* im2col_bf = (bf16_t*)alloc(sizeof(bf16_t) * (long)PROWS * DMODEL);
    float*  patch_tmp = (float*) alloc(sizeof(float)  * (long)PROWS * DMODEL);
    float*  hbuf      = (float*) alloc(sizeof(float)  * (long)NROWS * DMODEL);
    bf16_t* y_bf      = (bf16_t*)alloc(sizeof(bf16_t) * (long)NROWS * DMODEL);
    float*  qkvbuf    = (float*) alloc(sizeof(float)  * (long)NROWS * 3 * DMODEL);
    bf16_t* q_bf      = (bf16_t*)alloc(sizeof(bf16_t) * (long)NBH * NTOK * HDIM);
    bf16_t* ks_bf     = (bf16_t*)alloc(sizeof(bf16_t) * (long)NBH * NTOK * HDIM);
    bf16_t* k_bf      = (bf16_t*)alloc(sizeof(bf16_t) * (long)NBH * NTOK * HDIM);
    bf16_t* vT_bf     = (bf16_t*)alloc(sizeof(bf16_t) * (long)NBH * HDIM * NTOKP);
    float*  attnbuf   = (float*) alloc(sizeof(float)  * (long)NBH * NTOK * NTOK);
    bf16_t* attn_bf   = (bf16_t*)alloc(sizeof(bf16_t) * (long)NBH * NTOK * NTOKP);
    float*  qtbuf     = (float*) alloc(sizeof(float)  * (long)NBH * NTOK * NBUCKET);
    float*  ktbuf     = (float*) alloc(sizeof(float)  * (long)NBH * NTOK * NBUCKET);
    bf16_t* abkt_bf   = (bf16_t*)alloc(sizeof(bf16_t) * (long)NBH * NTOK * NBUCKP);
    float*  tmp_o     = (float*) alloc(sizeof(float)  * (long)NBH * NTOK * HDIM);
    bf16_t* o_bf      = (bf16_t*)alloc(sizeof(bf16_t) * (long)NROWS * DMODEL);
    bf16_t* ff_bf     = (bf16_t*)alloc(sizeof(bf16_t) * (long)NROWS * DFFN);
    bf16_t* cls_bf    = (bf16_t*)alloc(sizeof(bf16_t) * BATCHSZ * DMODEL);

    auto gemm = [&](const bf16_t* A, const bf16_t* Bt, float* C, bf16_t* Cb,
                    const float* bias, int Mr, int Nc, int K,
                    int lda, int ldb, int ldc, int ldcb,
                    long aB, long bB, long cB, int bMod, int batches, int flags) {
        dim3 g((Nc + BN - 1) / BN, (Mr + BM - 1) / BM, batches);
        hipLaunchKernelGGL(vit_gemm_bf16, g, dim3(256), 0, stream,
                           A, Bt, C, Cb, bias, Mr, Nc, K, lda, ldb, ldc, ldcb,
                           aB, bB, cB, bMod, flags);
    };
    auto cvt = [&](const float* s, bf16_t* d, long n) {
        hipLaunchKernelGGL(vit_cvt_bf16, dim3(4096), dim3(256), 0, stream, s, d, n);
    };
    auto wtrans = [&](const float* s, bf16_t* d, int K, int N, int ldo, int nb) {
        hipLaunchKernelGGL(vit_wtrans, dim3(4096), dim3(256), 0, stream,
                           s, d, K, N, ldo, (long)nb * K * N);
    };

    // ---- weight conversion: all GEMM B operands become Bt [N][K] bf16 ----
    wtrans(patch_w, wpatchT, DMODEL, DMODEL, DMODEL, 1);
    wtrans(qkv_w,   wqkvT,   DMODEL, 3 * DMODEL, DMODEL, NLAYER);
    wtrans(proj_w,  wprojT,  DMODEL, DMODEL, DMODEL, NLAYER);
    wtrans(fc1_w,   wfc1T,   DMODEL, DFFN, DMODEL, NLAYER);
    wtrans(fc2_w,   wfc2T,   DFFN, DMODEL, DFFN, NLAYER);
    wtrans(head_w,  wheadT,  DMODEL, 1000, DMODEL, 1);
    cvt(rpe_k, rpeK_bf, RPE_E);            // [lh][M][HD] is already Bt for qt
    cvt(rpe_q, rpeQ_bf, RPE_E);
    wtrans(rpe_v, rVT_bf, NBUCKET, HDIM, NBUCKP, NLAYER * NHEAD);  // -> [lh][64][232]

    // ---- patch embed ----
    {
        const long n = (long)PROWS * DMODEL;
        hipLaunchKernelGGL(vit_im2col, dim3((unsigned)((n + 255) / 256)), dim3(256), 0, stream, x, im2col_bf);
        gemm(im2col_bf, wpatchT, patch_tmp, nullptr, patch_b,
             PROWS, DMODEL, DMODEL, DMODEL, DMODEL, DMODEL, 0, 0, 0, 0, 1, 1, GF_WF32 | GF_BIAS);
        const long n2 = (long)NROWS * DMODEL;
        hipLaunchKernelGGL(vit_assemble_h, dim3((unsigned)((n2 + 255) / 256)), dim3(256), 0, stream,
                           patch_tmp, cls_tok, pos_emb, hbuf);
    }

    const long n_tok_el = (long)NROWS * DMODEL;
    const unsigned ew_blocks = (unsigned)((n_tok_el + 255) / 256);

    for (int l = 0; l < NLAYER; ++l) {
        const bf16_t* wqkv = wqkvT + (long)l * 3 * DMODEL * DMODEL;
        const bf16_t* wprj = wprojT + (long)l * DMODEL * DMODEL;
        const bf16_t* wf1  = wfc1T + (long)l * DFFN * DMODEL;
        const bf16_t* wf2  = wfc2T + (long)l * DMODEL * DFFN;
        const bf16_t* rK   = rpeK_bf + (long)l * NHEAD * NBUCKET * HDIM;
        const bf16_t* rQ   = rpeQ_bf + (long)l * NHEAD * NBUCKET * HDIM;
        const bf16_t* rVT  = rVT_bf  + (long)l * NHEAD * HDIM * NBUCKP;

        // LN1 -> bf16
        hipLaunchKernelGGL(vit_layernorm, dim3(NROWS), dim3(256), 0, stream,
                           hbuf, ln1_g + l * DMODEL, ln1_b + l * DMODEL, y_bf, 1);
        // qkv = y @ Wqkv
        gemm(y_bf, wqkv, qkvbuf, nullptr, nullptr, NROWS, 3 * DMODEL, DMODEL,
             DMODEL, DMODEL, 3 * DMODEL, 0, 0, 0, 0, 1, 1, GF_WF32);
        hipLaunchKernelGGL(vit_split_qkv, dim3(ew_blocks), dim3(256), 0, stream,
                           qkvbuf, q_bf, ks_bf, k_bf, vT_bf);
        // scores = q_scaled @ k^T  (Bt = k [n][d], per bh)
        gemm(q_bf, k_bf, attnbuf, nullptr, nullptr, NTOK, NTOK, HDIM,
             HDIM, HDIM, NTOK, 0,
             (long)NTOK * HDIM, (long)NTOK * HDIM, (long)NTOK * NTOK, NBH, NBH, GF_WF32);
        // qt = q_scaled @ rpe_k^T ; kt = k_scaled @ rpe_q^T (Bt shared per head)
        gemm(q_bf, rK, qtbuf, nullptr, nullptr, NTOK, NBUCKET, HDIM,
             HDIM, HDIM, NBUCKET, 0,
             (long)NTOK * HDIM, (long)NBUCKET * HDIM, (long)NTOK * NBUCKET, NHEAD, NBH, GF_WF32);
        gemm(ks_bf, rQ, ktbuf, nullptr, nullptr, NTOK, NBUCKET, HDIM,
             HDIM, HDIM, NBUCKET, 0,
             (long)NTOK * HDIM, (long)NBUCKET * HDIM, (long)NTOK * NBUCKET, NHEAD, NBH, GF_WF32);
        // softmax with RPE biases + bucket scatter
        hipLaunchKernelGGL(vit_softmax_rpe, dim3(NTOK, NBH), dim3(256), 0, stream,
                           attnbuf, qtbuf, ktbuf, rel_idx, attn_bf, abkt_bf);
        // out = attn @ v (Bt = v^T [d][n]) ; out += a_bucket @ rpe_v (Bt = rV^T)
        gemm(attn_bf, vT_bf, tmp_o, nullptr, nullptr, NTOK, HDIM, NTOK,
             NTOKP, NTOKP, HDIM, 0,
             (long)NTOK * NTOKP, (long)HDIM * NTOKP, (long)NTOK * HDIM, NBH, NBH, GF_WF32);
        gemm(abkt_bf, rVT, tmp_o, nullptr, nullptr, NTOK, HDIM, NBUCKET,
             NBUCKP, NBUCKP, HDIM, 0,
             (long)NTOK * NBUCKP, (long)HDIM * NBUCKP, (long)NTOK * HDIM, NHEAD, NBH,
             GF_WF32 | GF_ACC);
        hipLaunchKernelGGL(vit_reorder_o, dim3(ew_blocks), dim3(256), 0, stream, tmp_o, o_bf);
        // h += o @ Wproj + b
        gemm(o_bf, wprj, hbuf, nullptr, proj_b + l * DMODEL, NROWS, DMODEL, DMODEL,
             DMODEL, DMODEL, DMODEL, 0, 0, 0, 0, 1, 1, GF_WF32 | GF_ACC | GF_BIAS);
        // LN2 -> bf16
        hipLaunchKernelGGL(vit_layernorm, dim3(NROWS), dim3(256), 0, stream,
                           hbuf, ln2_g + l * DMODEL, ln2_b + l * DMODEL, y_bf, 1);
        // ff = gelu(y @ W1 + b1) -> bf16
        gemm(y_bf, wf1, nullptr, ff_bf, fc1_b + l * DFFN, NROWS, DFFN, DMODEL,
             DMODEL, DMODEL, 0, DFFN, 0, 0, 0, 1, 1, GF_BIAS | GF_GELU | GF_WBF);
        // h += ff @ W2 + b2
        gemm(ff_bf, wf2, hbuf, nullptr, fc2_b + l * DMODEL, NROWS, DMODEL, DFFN,
             DFFN, DFFN, DMODEL, 0, 0, 0, 0, 1, 1, GF_WF32 | GF_ACC | GF_BIAS);
    }

    // ---- final LN on cls rows only + classifier head ----
    hipLaunchKernelGGL(vit_layernorm, dim3(BATCHSZ), dim3(256), 0, stream,
                       hbuf, norm_g, norm_b, cls_bf, NTOK);
    gemm(cls_bf, wheadT, out, nullptr, head_b, BATCHSZ, 1000, DMODEL,
         DMODEL, DMODEL, 1000, 0, 0, 0, 0, 1, 1, GF_WF32 | GF_BIAS);
}
